// GNN_56994216018053
// MI455X (gfx1250) — matrix operand order repaired and checked by reading, not compiled, not run on hardware
//
#include <hip/hip_runtime.h>

// ---------------- problem constants ----------------
#define NNODES 100000
#define NEDGES 3200000
#define NGRAPHS 64
#define NCLASSES 10

typedef __attribute__((ext_vector_type(16))) __bf16 v16bf;
typedef __attribute__((ext_vector_type(8)))  float  v8f;

union Frag { uint4 q[2]; v16bf v; };

// round-to-nearest-even f32->bf16
__device__ __forceinline__ unsigned short f32_to_bf16(float f) {
    unsigned int u = __float_as_uint(f);
    u += 0x7FFFu + ((u >> 16) & 1u);
    return (unsigned short)(u >> 16);
}
// two at a time packed into a dword
__device__ __forceinline__ unsigned int pack2_bf16(float a, float b) {
    unsigned int ua = __float_as_uint(a);
    unsigned int ub = __float_as_uint(b);
    ua += 0x7FFFu + ((ua >> 16) & 1u);
    ub += 0x7FFFu + ((ub >> 16) & 1u);
    return (ua >> 16) | (ub & 0xFFFF0000u);
}
__device__ __forceinline__ float bf16lo(unsigned int u) { return __uint_as_float(u << 16); }
__device__ __forceinline__ float bf16hi(unsigned int u) { return __uint_as_float(u & 0xFFFF0000u); }

// ---------------- degree / normalization ----------------
__global__ __launch_bounds__(256) void deg_kernel(const int* __restrict__ src,
                                                  const int* __restrict__ dst,
                                                  float* __restrict__ od,
                                                  float* __restrict__ id, int E) {
    int e = blockIdx.x * 256 + threadIdx.x;
    if (e < E) {
        atomicAdd(&od[src[e]], 1.0f);
        atomicAdd(&id[dst[e]], 1.0f);
    }
}

__global__ __launch_bounds__(256) void invsqrt_kernel(float* __restrict__ a,
                                                      float* __restrict__ b, int n) {
    int i = blockIdx.x * 256 + threadIdx.x;
    if (i < n) {
        a[i] = rsqrtf(fmaxf(a[i], 1.0f));
        b[i] = rsqrtf(fmaxf(b[i], 1.0f));
    }
}

// ---------------- f32 -> packed bf16 conversion (vectorized) ----------------
__global__ __launch_bounds__(256) void f2b_kernel(const float* __restrict__ X,
                                                  unsigned short* __restrict__ Y, size_t n4) {
    size_t i = (size_t)blockIdx.x * 256 + threadIdx.x;
    if (i >= n4) return;
    float4 v = ((const float4*)X)[i];
    uint2 u;
    u.x = pack2_bf16(v.x, v.y);
    u.y = pack2_bf16(v.z, v.w);
    ((uint2*)Y)[i] = u;
}

// ---------------- SpMM: Yf32[dst] += bf16 X[src] * (ss ? ss[src] : 1) ----------------
// one wave per edge; bf16 gather halves the roofline-limiting read traffic,
// f32 atomics keep accumulation precision
__global__ __launch_bounds__(256) void spmm_kernel(const unsigned short* __restrict__ X,
                                                   const int* __restrict__ src,
                                                   const int* __restrict__ dst,
                                                   const float* __restrict__ ss,
                                                   float* __restrict__ Y,
                                                   int E, int F) {
    int wid  = threadIdx.x >> 5;
    int lane = threadIdx.x & 31;
    long e = (long)blockIdx.x * 8 + wid;
    if (e >= E) return;
    int s = src[e], d = dst[e];
    float sc = ss ? ss[s] : 1.0f;
    const uint4* xp = (const uint4*)(X + (size_t)s * F);
    float* yp = Y + (size_t)d * F;
    int nq = F >> 3;                      // 8 bf16 per uint4
    for (int q = lane; q < nq; q += 32) {
        uint4 u = xp[q];
        int f = q * 8;
        atomicAdd(yp + f + 0, bf16lo(u.x) * sc);
        atomicAdd(yp + f + 1, bf16hi(u.x) * sc);
        atomicAdd(yp + f + 2, bf16lo(u.y) * sc);
        atomicAdd(yp + f + 3, bf16hi(u.y) * sc);
        atomicAdd(yp + f + 4, bf16lo(u.z) * sc);
        atomicAdd(yp + f + 5, bf16hi(u.z) * sc);
        atomicAdd(yp + f + 6, bf16lo(u.w) * sc);
        atomicAdd(yp + f + 7, bf16hi(u.w) * sc);
    }
}

// ---- post-aggregation epilogue: Yb = bf16(relu(Yf*ins[row] + b[col])) ----
__global__ __launch_bounds__(256) void sbr_kernel(const float* __restrict__ Yf,
                                                  unsigned short* __restrict__ Yb,
                                                  const float* __restrict__ ins,
                                                  const float* __restrict__ b,
                                                  size_t total, int fmask, int shift) {
    size_t i = (size_t)blockIdx.x * 256 + threadIdx.x;
    if (i >= total) return;
    int row = (int)(i >> shift);
    int col = (int)(i & fmask);
    float x = Yf[i] * ins[row] + b[col];
    Yb[i] = f32_to_bf16(fmaxf(x, 0.0f));
}

// ---------------- WMMA bf16 GEMM ----------------
// C_bf16[M,N] = relu?( (rs_in.*A)[M,K] @ W[K,N] .* rs_out + bias )
// 256 threads = 8 waves (4 along M x 2 along N); block tile 128x128, K-step 32.
// Software pipelined: next K-tile loads to registers while WMMAs run from LDS.
// ABF16: A is already bf16 -> staging is a raw b128 copy (zero conversion VALU).
#define BM 128
#define BN 128
#define BK 32
#define LDA 40   // shorts, padded
#define LDB 40

template <bool ABF16>
__global__ __launch_bounds__(256) void wmma_gemm(const void* __restrict__ Av,
                                                 const float* __restrict__ W,
                                                 unsigned short* __restrict__ C,
                                                 int M, int K, int N,
                                                 const float* __restrict__ rs_in,
                                                 const float* __restrict__ rs_out,
                                                 const float* __restrict__ bias,
                                                 int relu) {
    __shared__ __align__(16) unsigned short sA[BM * LDA];
    __shared__ __align__(16) unsigned short sB[BN * LDB];

    const int tid  = threadIdx.x;
    const int lane = tid & 31;
    const int wid  = tid >> 5;
    const int wm   = wid & 3;        // wave row 0..3  (32 rows each)
    const int wn   = wid >> 2;       // wave col 0..1  (64 cols each)
    const int block_m = blockIdx.x * BM;
    const int block_n = blockIdx.y * BN;

    v8f acc[2][4] = {};

    // ---- W staging: n-major reads, packed b64 transposed store ----
    const int nn = tid & 127;
    const int kq = (tid >> 7) * 4;       // 0 or 4
    const int gnW = block_n + nn;
    const bool wOk = gnW < N;
    const float* wPtr = W + (wOk ? gnW : 0);
    const float wSc = wOk ? 1.0f : 0.0f;
    float wreg[4][4];

    // ---- A staging descriptors (branch-free edge handling) ----
    const float*          Af = (const float*)Av;
    const unsigned short* Ab = (const unsigned short*)Av;
    // f32 path: rows arow+{0,32,64,96}, cols acol..acol+3
    const int arow = tid >> 3;
    const int acol = (tid & 7) * 4;
    const float* aPtrF[4]; float aScF[4]; float4 aregF[4];
    // bf16 path: 2 b128 quads per thread
    int brow[2], bcol[2]; const unsigned short* aPtrB[2]; bool aOkB[2]; uint4 aregB[2];

    if constexpr (ABF16) {
#pragma unroll
        for (int i = 0; i < 2; ++i) {
            int q = tid + i * 256;       // 0..511
            brow[i] = q >> 2;            // 0..127
            bcol[i] = (q & 3) * 8;       // 0,8,16,24 (shorts)
            int gr = block_m + brow[i];
            aOkB[i] = gr < M;
            aPtrB[i] = Ab + (size_t)(aOkB[i] ? gr : 0) * K + bcol[i];
        }
    } else {
#pragma unroll
        for (int i = 0; i < 4; ++i) {
            int gr = block_m + arow + i * 32;
            bool ok = gr < M;
            aPtrF[i] = Af + (size_t)(ok ? gr : 0) * K + acol;
            aScF[i]  = ok ? (rs_in ? rs_in[gr] : 1.0f) : 0.0f;
        }
    }

    auto loadTiles = [&](int kb) {
        if constexpr (ABF16) {
#pragma unroll
            for (int i = 0; i < 2; ++i) {
                uint4 v = *(const uint4*)(aPtrB[i] + kb);
                if (!aOkB[i]) { v.x = 0; v.y = 0; v.z = 0; v.w = 0; }
                aregB[i] = v;
            }
        } else {
#pragma unroll
            for (int i = 0; i < 4; ++i) {
                float4 v = *(const float4*)(aPtrF[i] + kb);
                float s = aScF[i];
                v.x *= s; v.y *= s; v.z *= s; v.w *= s;
                aregF[i] = v;
            }
        }
#pragma unroll
        for (int j = 0; j < 4; ++j) {
            int k = kq + j * 8;
#pragma unroll
            for (int c = 0; c < 4; ++c)
                wreg[j][c] = wPtr[(size_t)(kb + k + c) * N] * wSc;
        }
    };
    auto storeTiles = [&]() {
        if constexpr (ABF16) {
#pragma unroll
            for (int i = 0; i < 2; ++i)
                *(uint4*)&sA[brow[i] * LDA + bcol[i]] = aregB[i];
        } else {
#pragma unroll
            for (int i = 0; i < 4; ++i) {
                uint2 u;
                u.x = pack2_bf16(aregF[i].x, aregF[i].y);
                u.y = pack2_bf16(aregF[i].z, aregF[i].w);
                *(uint2*)&sA[(arow + i * 32) * LDA + acol] = u;
            }
        }
#pragma unroll
        for (int j = 0; j < 4; ++j) {
            uint2 u;
            u.x = pack2_bf16(wreg[j][0], wreg[j][1]);
            u.y = pack2_bf16(wreg[j][2], wreg[j][3]);
            *(uint2*)&sB[nn * LDB + kq + j * 8] = u;
        }
    };

    loadTiles(0);
    for (int kb = 0; kb < K; kb += BK) {
        storeTiles();
        __syncthreads();
        if (kb + BK < K) loadTiles(kb + BK);   // prefetch next tile during compute

        // ---- fragment loads per ISA VGPR layouts + 2x4 WMMAs ----
        const int r16 = lane & 15;
        const int khA = (lane >> 4) * 8;   // A: K split {0..7,16..23} (+8 hi lanes)
        const int kB  = (lane >> 4) * 8;   // B: K split {0..7,16..23} (+8 hi lanes)
        Frag a[2], b[4];
#pragma unroll
        for (int tm = 0; tm < 2; ++tm) {
            const unsigned short* p = &sA[(wm * 32 + tm * 16 + r16) * LDA + khA];
            a[tm].q[0] = *(const uint4*)p;
            a[tm].q[1] = *(const uint4*)(p + 16);
        }
#pragma unroll
        for (int tn = 0; tn < 4; ++tn) {
            const unsigned short* p = &sB[(wn * 64 + tn * 16 + r16) * LDB + kB];
            b[tn].q[0] = *(const uint4*)p;
            b[tn].q[1] = *(const uint4*)(p + 16);
        }
#pragma unroll
        for (int tm = 0; tm < 2; ++tm)
#pragma unroll
            for (int tn = 0; tn < 4; ++tn)
                acc[tm][tn] = __builtin_amdgcn_wmma_f32_16x16x32_bf16(
                    false, a[tm].v, false, b[tn].v, (short)0, acc[tm][tn], false, false);
        __syncthreads();
    }

    // ---- epilogue: rs_out row scale, bias, relu, bf16 store ----
    const int col16  = lane & 15;
    const int rowoff = (lane >> 4) * 8;
#pragma unroll
    for (int tm = 0; tm < 2; ++tm) {
        float rsv[8];
#pragma unroll
        for (int v = 0; v < 8; ++v) {
            int grow = block_m + wm * 32 + tm * 16 + rowoff + v;
            rsv[v] = rs_out ? ((grow < M) ? rs_out[grow] : 0.0f) : 1.0f;
        }
#pragma unroll
        for (int tn = 0; tn < 4; ++tn) {
            int gcol = block_n + wn * 64 + tn * 16 + col16;
            if (gcol >= N) continue;
            float bv = bias ? bias[gcol] : 0.0f;
#pragma unroll
            for (int v = 0; v < 8; ++v) {
                int grow = block_m + wm * 32 + tm * 16 + rowoff + v;
                if (grow < M) {
                    float x = acc[tm][tn][v] * rsv[v] + bv;
                    if (relu) x = fmaxf(x, 0.0f);
                    C[(size_t)grow * N + gcol] = f32_to_bf16(x);
                }
            }
        }
    }
}

// ---------------- mean pooling + classifier ----------------
__global__ __launch_bounds__(256) void pool_kernel(const unsigned short* __restrict__ h,
                                                   const int* __restrict__ gid,
                                                   float* __restrict__ gsum,
                                                   float* __restrict__ gcnt, int n) {
    int idx  = blockIdx.x * 256 + threadIdx.x;
    int node = idx >> 5;
    int f    = idx & 31;
    if (node < n) {
        int g = gid[node];
        float v = __uint_as_float((unsigned int)h[(size_t)node * 32 + f] << 16);
        atomicAdd(&gsum[g * 32 + f], v);
        if (f == 0) atomicAdd(&gcnt[g], 1.0f);
    }
}

__global__ void cls_kernel(const float* __restrict__ gsum, const float* __restrict__ gcnt,
                           const float* __restrict__ Wc, const float* __restrict__ bc,
                           float* __restrict__ out) {
    int g = blockIdx.x, lane = threadIdx.x;
    __shared__ float hg[32];
    float c = fmaxf(gcnt[g], 1.0f);
    hg[lane] = gsum[g * 32 + lane] / c;
    __syncthreads();
    if (lane < NCLASSES) {
        float acc = bc[lane];
#pragma unroll
        for (int k = 0; k < 32; ++k) acc += hg[k] * Wc[k * NCLASSES + lane];
        out[g * NCLASSES + lane] = acc;
    }
}

// ---------------- host orchestration ----------------
extern "C" void kernel_launch(void* const* d_in, const int* in_sizes, int n_in,
                              void* d_out, int out_size, void* d_ws, size_t ws_size,
                              hipStream_t stream) {
    (void)in_sizes; (void)n_in; (void)out_size;
    const float* h0  = (const float*)d_in[0];
    const int*   src = (const int*)d_in[1];
    const int*   dst = (const int*)d_in[2];
    const int*   gid = (const int*)d_in[3];
    const float* Ws[6], *bs[6];
    for (int i = 0; i < 6; ++i) { Ws[i] = (const float*)d_in[4 + 2 * i]; bs[i] = (const float*)d_in[5 + 2 * i]; }
    const float* Wc = (const float*)d_in[16];
    const float* bc = (const float*)d_in[17];
    float* out = (float*)d_out;

    // workspace carve-up
    const size_t AGGF = (size_t)NNODES * 512;        // f32 aggregation buffer (max width 512)
    const size_t SZ_P = (size_t)NNODES * 1024;       // bf16 h (widest)
    const size_t SZ_Q = (size_t)NNODES * 512;        // bf16 h0 / GEMM outputs
    float*          aggF = (float*)d_ws;
    unsigned short* P    = (unsigned short*)(aggF + AGGF);
    unsigned short* Q    = P + SZ_P;
    float* outs = (float*)(Q + SZ_Q);                // out_inv_sqrt
    float* ins  = outs + NNODES;                     // in_inv_sqrt
    float* gsum = ins + NNODES;                      // 64*32
    float* gcnt = gsum + NGRAPHS * 32;               // 64
    size_t needed = ((size_t)(gcnt + NGRAPHS) - (size_t)d_ws);
    if (ws_size < needed) return;

    const int dims[7] = {512, 1024, 512, 256, 128, 64, 32};

    // degrees -> inverse sqrt norms (outs/ins are contiguous)
    hipMemsetAsync(outs, 0, (size_t)2 * NNODES * sizeof(float), stream);
    deg_kernel<<<(NEDGES + 255) / 256, 256, 0, stream>>>(src, dst, outs, ins, NEDGES);
    invsqrt_kernel<<<(NNODES + 255) / 256, 256, 0, stream>>>(outs, ins, NNODES);

    // h0 (f32) -> Q (bf16)
    {
        size_t n4 = (size_t)NNODES * 512 / 4;
        f2b_kernel<<<(unsigned)((n4 + 255) / 256), 256, 0, stream>>>(h0, Q, n4);
    }

    // ---- layer 1 (512 -> 1024): aggregate first, then GEMM(+rs_in, bias, relu) ----
    hipMemsetAsync(aggF, 0, (size_t)NNODES * 512 * sizeof(float), stream);
    spmm_kernel<<<(NEDGES + 7) / 8, 256, 0, stream>>>(Q, src, dst, outs, aggF, NEDGES, 512);
    {
        dim3 grid((NNODES + BM - 1) / BM, (1024 + BN - 1) / BN);
        wmma_gemm<false><<<grid, 256, 0, stream>>>(aggF, Ws[0], P, NNODES, 512, 1024,
                                                   ins, nullptr, bs[0], 1);
    }

    // ---- layers 2..6 (shrinking): GEMM(rs_out=outs) -> aggregate -> scale+bias+relu ----
    for (int L = 1; L < 6; ++L) {
        int fi = dims[L], fo = dims[L + 1];
        dim3 grid((NNODES + BM - 1) / BM, (fo + BN - 1) / BN);
        wmma_gemm<true><<<grid, 256, 0, stream>>>(P, Ws[L], Q, NNODES, fi, fo,
                                                  nullptr, outs, nullptr, 0);
        hipMemsetAsync(aggF, 0, (size_t)NNODES * fo * sizeof(float), stream);
        spmm_kernel<<<(NEDGES + 7) / 8, 256, 0, stream>>>(Q, src, dst, nullptr, aggF,
                                                          NEDGES, fo);
        int shift = 0; while ((1 << shift) < fo) ++shift;   // fo is a power of two
        size_t tot = (size_t)NNODES << shift;
        sbr_kernel<<<(unsigned)((tot + 255) / 256), 256, 0, stream>>>(aggF, P, ins, bs[L],
                                                                      tot, fo - 1, shift);
    }

    // ---- mean pooling + classifier (h = P, bf16, F=32) ----
    hipMemsetAsync(gsum, 0, (size_t)(NGRAPHS * 32 + NGRAPHS) * sizeof(float), stream);
    pool_kernel<<<(NNODES * 32 + 255) / 256, 256, 0, stream>>>(P, gid, gsum, gcnt, NNODES);
    cls_kernel<<<NGRAPHS, 32, 0, stream>>>(gsum, gcnt, Wc, bc, out);
}
